// Attention_84447646974661
// MI455X (gfx1250) — matrix operand order repaired and checked by reading, not compiled, not run on hardware
//
#include <hip/hip_runtime.h>
#include <hip/hip_bf16.h>

// ---------------------------------------------------------------------------
// Problem constants (from reference): B=2, S=2048, DM=1024, H=16, KH=4, HD=64
// ---------------------------------------------------------------------------
constexpr int Bc  = 2;
constexpr int Sc  = 2048;
constexpr int DMc = 1024;
constexpr int Hc  = 16;
constexpr int KHc = 4;
constexpr int HDc = 64;
constexpr int NREP = Hc / KHc;

// softmax scale folded with log2(e): scores kept in log2 domain so the
// hardware base-2 v_exp_f32 needs no extra multiply.
constexpr float SCL2 = 0.125f * 1.4426950408889634f;

typedef __attribute__((ext_vector_type(8)))  _Float16 half8_t;
typedef __attribute__((ext_vector_type(16))) _Float16 v16h;
typedef __attribute__((ext_vector_type(8)))  float    v8f;

__device__ __forceinline__ half8_t ld8(const _Float16* p) {
    return *reinterpret_cast<const half8_t*>(p);
}
__device__ __forceinline__ v16h ld16(const _Float16* p) {
    return *reinterpret_cast<const v16h*>(p);
}
__device__ __forceinline__ v16h cat16(half8_t a, half8_t b) {
    return __builtin_shufflevector(a, b, 0,1,2,3,4,5,6,7,8,9,10,11,12,13,14,15);
}
__device__ __forceinline__ v8f wmma16(v16h a, v16h b, v8f c) {
    // D = A(16x32 f16) * B(32x16 f16) + C(16x16 f32)
    return __builtin_amdgcn_wmma_f32_16x16x32_f16(false, a, false, b,
                                                  (short)0, c, false, false);
}

// Raw hardware base-2 exponential (v_exp_f32). Avoids libm's subnormal-range
// fixup (cmp/cndmask/ldexp chain); deep underflow flushes to 0, which is the
// desired behavior for masked lanes at -1e30.
__device__ __forceinline__ float fast_exp2(float x) {
#if __has_builtin(__builtin_amdgcn_exp2f)
    return __builtin_amdgcn_exp2f(x);
#else
    return exp2f(x);
#endif
}

// ---------------------------------------------------------------------------
// 16-lane reductions via DPP16 rotate-reduce (pure VALU, no LDS round trip):
//   quad_perm xor1, quad_perm xor2, row_ror:4, row_ror:8
// stays inside each 16-lane row, matching the WMMA C/D N-striping.
// ---------------------------------------------------------------------------
template <int CTRL>
__device__ __forceinline__ float dpp_f32(float v) {
    int r = __builtin_amdgcn_update_dpp(0, __builtin_bit_cast(int, v),
                                        CTRL, 0xF, 0xF, true);
    return __builtin_bit_cast(float, r);
}
__device__ __forceinline__ float rowmax16(float v) {
    v = fmaxf(v, dpp_f32<0x0B1>(v));   // quad_perm [1,0,3,2]
    v = fmaxf(v, dpp_f32<0x04E>(v));   // quad_perm [2,3,0,1]
    v = fmaxf(v, dpp_f32<0x124>(v));   // row_ror:4
    v = fmaxf(v, dpp_f32<0x128>(v));   // row_ror:8
    return v;
}
__device__ __forceinline__ float rowsum16(float v) {
    v += dpp_f32<0x0B1>(v);
    v += dpp_f32<0x04E>(v);
    v += dpp_f32<0x124>(v);
    v += dpp_f32<0x128>(v);
    return v;
}

// ---------------------------------------------------------------------------
// Elementwise / layout prep kernels
// ---------------------------------------------------------------------------
__global__ void k_f32_to_f16(const float* __restrict__ in,
                             _Float16* __restrict__ out, int n) {
    int i = blockIdx.x * blockDim.x + threadIdx.x;
    if (i < n) out[i] = (_Float16)in[i];
}

// w: [K][N] f32 -> wt: [N][K] f16. LDS-tiled 32x32 transpose: coalesced
// global reads AND writes (naive version pays ~32x cacheline amplification).
// grid = (N/32, K/32), block = (32, 8)
__global__ void k_transpose_f16(const float* __restrict__ w,
                                _Float16* __restrict__ wt, int K, int N) {
    __shared__ float tile[32][33];
    const int n0 = blockIdx.x * 32;
    const int k0 = blockIdx.y * 32;
    const int tx = threadIdx.x, ty = threadIdx.y;
#pragma unroll
    for (int i = 0; i < 4; ++i)
        tile[ty + i * 8][tx] = w[(size_t)(k0 + ty + i * 8) * N + n0 + tx];
    __syncthreads();
#pragma unroll
    for (int i = 0; i < 4; ++i)
        wt[(size_t)(n0 + ty + i * 8) * K + k0 + tx] =
            (_Float16)tile[tx][ty + i * 8];
}

// in-place RoPE on buf laid out [B, S, nh, HD] (f16); each thread owns a pair
__global__ void k_rope(_Float16* __restrict__ buf,
                       const float* __restrict__ cosb,
                       const float* __restrict__ sinb, int nh, int total) {
    int i = blockIdx.x * blockDim.x + threadIdx.x;   // total = B*S*nh*32
    if (i >= total) return;
    int d    = i % (HDc / 2);
    int rest = i / (HDc / 2);                        // [B,S,nh] flat
    int s    = (rest / nh) % Sc;
    size_t base = (size_t)rest * HDc;
    float c0 = cosb[s * HDc + d],        s0 = sinb[s * HDc + d];
    float c1 = cosb[s * HDc + d + 32],   s1 = sinb[s * HDc + d + 32];
    float u = (float)buf[base + d];
    float w = (float)buf[base + d + 32];
    buf[base + d]      = (_Float16)(u * c0 - w * s0);
    buf[base + d + 32] = (_Float16)(w * c1 + u * s1);
}

// v: [B*S][KH*HD] f16 -> vt: [B][KH][HD][S] f16. LDS-tiled (s,d) transpose.
// grid = (HD/32, S/32, B*KH), block = (32, 8)
__global__ void k_vtrans(const _Float16* __restrict__ v,
                         _Float16* __restrict__ vt) {
    __shared__ _Float16 tile[32][33];
    const int d0 = blockIdx.x * 32;
    const int s0 = blockIdx.y * 32;
    const int bk = blockIdx.z;             // b*KH + kh
    const int b  = bk / KHc, kh = bk % KHc;
    const int tx = threadIdx.x, ty = threadIdx.y;
    const _Float16* src = v + (size_t)(b * Sc) * (KHc * HDc) + kh * HDc;
#pragma unroll
    for (int i = 0; i < 4; ++i)
        tile[ty + i * 8][tx] =
            src[(size_t)(s0 + ty + i * 8) * (KHc * HDc) + d0 + tx];
    __syncthreads();
    _Float16* dst = vt + ((size_t)bk * HDc) * Sc;
#pragma unroll
    for (int i = 0; i < 4; ++i)
        dst[(size_t)(d0 + ty + i * 8) * Sc + s0 + tx] = tile[tx][ty + i * 8];
}

// ---------------------------------------------------------------------------
// WMMA GEMM:  C[M][N] = A[M][K] * Bt[N][K]^T   (f16 in, f32 acc, OutT out)
// block = 128 threads (4 waves); each wave owns a 32x64 output tile so the
// B operand is reused by two row tiles (8 WMMAs per 4 B-loads per k-step).
// grid = (N/64, M/128)
// ---------------------------------------------------------------------------
template <typename OutT>
__global__ void k_gemm_wmma(const _Float16* __restrict__ A,
                            const _Float16* __restrict__ Bt,
                            OutT* __restrict__ C, int M, int N, int K) {
    const int lane = threadIdx.x & 31;
    const int wave = threadIdx.x >> 5;
    const int n    = lane & 15;
    const int hi   = lane >> 4;
    const int row0 = blockIdx.y * 128 + wave * 32;
    const int col0 = blockIdx.x * 64;

    const _Float16* arow0 = A + (size_t)(row0 + n) * K;
    const _Float16* arow1 = arow0 + (size_t)16 * K;
    v8f acc0[4] = {}, acc1[4] = {};

    for (int kk = 0; kk < K; kk += 32) {
        v16h a0 = ld16(arow0 + kk + hi * 16);
        v16h a1 = ld16(arow1 + kk + hi * 16);
        __builtin_prefetch(arow0 + kk + 64, 0, 3);
        __builtin_prefetch(arow1 + kk + 64, 0, 3);
#pragma unroll
        for (int t = 0; t < 4; ++t) {
            const _Float16* bp = Bt + (size_t)(col0 + t * 16 + n) * K + kk;
            v16h b = ld16(bp + hi * 16);
            __builtin_prefetch(bp + 64, 0, 3);
            acc0[t] = wmma16(a0, b, acc0[t]);
            acc1[t] = wmma16(a1, b, acc1[t]);
        }
    }
#pragma unroll
    for (int t = 0; t < 4; ++t)
#pragma unroll
        for (int r = 0; r < 8; ++r) {
            int col = col0 + t * 16 + n;
            C[(size_t)(row0 + r + hi * 8) * N + col]      = (OutT)acc0[t][r];
            C[(size_t)(row0 + 16 + r + hi * 8) * N + col] = (OutT)acc1[t][r];
        }
}

// ---------------------------------------------------------------------------
// Flash attention building block: one 32-key block for a 16-row Q tile.
// MASKED=false: block is strictly below the causal diagonal (no compares).
// All softmax state is in the log2 domain (hardware exp is base-2).
// ---------------------------------------------------------------------------
template <bool MASKED>
__device__ __forceinline__ void attn_block(
    int kb, int qbase, int n, int hi,
    const _Float16* __restrict__ kbase, const _Float16* __restrict__ vbase,
    const v16h& aq0, const v16h& aq1,
    float (&rm)[8], float (&rlp)[8], v8f (&oacc)[4],
    _Float16 (&pl)[16][32])
{
    // ---- S = Q * K^T (two 16x16 score tiles covering 32 keys) ----
    const _Float16* k0 = kbase + (size_t)(kb + n)      * (KHc * HDc);
    const _Float16* k1 = kbase + (size_t)(kb + 16 + n) * (KHc * HDc);
    v8f s0 = {}, s1 = {};
    s0 = wmma16(aq0, ld16(k0 + hi * 16),      s0);
    s0 = wmma16(aq1, ld16(k0 + 32 + hi * 16), s0);
    s1 = wmma16(aq0, ld16(k1 + hi * 16),      s1);
    s1 = wmma16(aq1, ld16(k1 + 32 + hi * 16), s1);

    // prefetch next key block (global_prefetch_b8)
    __builtin_prefetch(k0 + 32 * (KHc * HDc), 0, 3);
    __builtin_prefetch(k1 + 32 * (KHc * HDc), 0, 3);

    // ---- online softmax; rows live in VGPR r, N striped over 16 lanes ----
#pragma unroll
    for (int r = 0; r < 8; ++r) {
        float x0 = s0[r] * SCL2;
        float x1 = s1[r] * SCL2;
        if (MASKED) {
            int rowg = qbase + r + hi * 8;
            if (kb + n      > rowg) x0 = -1e30f;
            if (kb + 16 + n > rowg) x1 = -1e30f;
        }
        float lm = rowmax16(fmaxf(x0, x1));
        float nm = fmaxf(rm[r], lm);
        float f  = fast_exp2(rm[r] - nm);
        rm[r] = nm;
        float e0 = fast_exp2(x0 - nm);
        float e1 = fast_exp2(x1 - nm);
        rlp[r] = rlp[r] * f + (e0 + e1);   // per-lane partial row sum
#pragma unroll
        for (int t = 0; t < 4; ++t) oacc[t][r] *= f;
        pl[r + hi * 8][n]      = (_Float16)e0;
        pl[r + hi * 8][16 + n] = (_Float16)e1;
    }

    // ---- re-read P in A-operand layout (same wave; dscnt ordered) ----
    v16h ap = cat16(ld8(&pl[n][hi * 8]), ld8(&pl[n][16 + hi * 8]));

    // ---- O += P * V (B-operand in table order from transposed V) ----
#pragma unroll
    for (int t = 0; t < 4; ++t) {
        const _Float16* vp = vbase + (size_t)(t * 16 + n) * Sc + kb;
        v16h bv = cat16(ld8(vp + hi * 8), ld8(vp + 16 + hi * 8));
        oacc[t] = wmma16(ap, bv, oacc[t]);
    }
}

// ---------------------------------------------------------------------------
// Causal flash attention, one wave per 16-row Q tile, key blocks of 32.
// q  : [B, S, H, HD]  f16 (RoPE'd)
// kb : [B, S, KH, HD] f16 (RoPE'd)
// vt : [B, KH, HD, S] f16
// o  : [B, S, H*HD]   f16
// grid = (S/64, H, B), block = 128 (4 waves, 4 q-tiles)
// ---------------------------------------------------------------------------
__global__ void k_flash_attn(const _Float16* __restrict__ q,
                             const _Float16* __restrict__ kbuf,
                             const _Float16* __restrict__ vt,
                             _Float16* __restrict__ o) {
    __shared__ _Float16 plds[4][16][32];   // P tile staging per wave

    const int lane = threadIdx.x & 31;
    const int wv   = threadIdx.x >> 5;
    const int n    = lane & 15;
    const int hi   = lane >> 4;
    const int b    = blockIdx.z;
    const int h    = blockIdx.y;
    const int kvh  = h / NREP;
    const int qbase = (blockIdx.x * 4 + wv) * 16;

    // A-operand Q tile (16 x 64 split into two 16x32 K-chunks)
    const _Float16* qrow =
        q + ((size_t)(b * Sc + qbase + n) * (Hc * HDc) + h * HDc);
    v16h aq0 = ld16(qrow + hi * 16);
    v16h aq1 = ld16(qrow + 32 + hi * 16);

    const _Float16* kbase = kbuf + (size_t)b * Sc * (KHc * HDc) + kvh * HDc;
    const _Float16* vbase = vt + ((size_t)(b * KHc + kvh) * HDc) * Sc;

    float rm[8], rlp[8];
#pragma unroll
    for (int r = 0; r < 8; ++r) { rm[r] = -1e30f; rlp[r] = 0.0f; }
    v8f oacc[4] = {};

    // full blocks strictly below the diagonal: no masking, no compares
    int kb = 0;
    for (; kb + 32 <= qbase; kb += 32)
        attn_block<false>(kb, qbase, n, hi, kbase, vbase, aq0, aq1,
                          rm, rlp, oacc, plds[wv]);
    // exactly one diagonal block with per-element causal mask
    attn_block<true>(kb, qbase, n, hi, kbase, vbase, aq0, aq1,
                     rm, rlp, oacc, plds[wv]);

    // ---- normalize (single deferred row-sum reduction) and store ----
#pragma unroll
    for (int r = 0; r < 8; ++r) {
        float inv = 1.0f / rowsum16(rlp[r]);
        int rowg  = qbase + r + hi * 8;
#pragma unroll
        for (int t = 0; t < 4; ++t) {
            o[(size_t)(b * Sc + rowg) * (Hc * HDc) + h * HDc + t * 16 + n] =
                (_Float16)(oacc[t][r] * inv);
        }
    }
}

// ---------------------------------------------------------------------------
// Host-side launch
// ---------------------------------------------------------------------------
static inline size_t align256(size_t x) { return (x + 255) & ~(size_t)255; }

extern "C" void kernel_launch(void* const* d_in, const int* in_sizes, int n_in,
                              void* d_out, int out_size, void* d_ws, size_t ws_size,
                              hipStream_t stream) {
    const float* x    = (const float*)d_in[0];  // [B,S,DM]
    const float* cosb = (const float*)d_in[1];  // [S,HD]
    const float* sinb = (const float*)d_in[2];  // [S,HD]
    const float* wq   = (const float*)d_in[3];  // [DM, H*HD]
    const float* wk   = (const float*)d_in[4];  // [DM, KH*HD]
    const float* wv   = (const float*)d_in[5];  // [DM, KH*HD]
    const float* wo   = (const float*)d_in[6];  // [H*HD, DM]
    float* out = (float*)d_out;                 // [B,S,DM]

    const int M   = Bc * Sc;          // 4096 token rows
    const int NQ  = Hc * HDc;         // 1024
    const int NKV = KHc * HDc;        // 256

    // workspace carve-up (f16 regions, 256B aligned). obuf overlays xh
    // (xh is dead after the V projection; obuf is written later).
    char* ws = (char*)d_ws;
    size_t off = 0;
    _Float16* xh   = (_Float16*)(ws + off); off = align256(off + (size_t)M * DMc * 2);
    _Float16* wqT  = (_Float16*)(ws + off); off = align256(off + (size_t)NQ  * DMc * 2);
    _Float16* wkT  = (_Float16*)(ws + off); off = align256(off + (size_t)NKV * DMc * 2);
    _Float16* wvT  = (_Float16*)(ws + off); off = align256(off + (size_t)NKV * DMc * 2);
    _Float16* woT  = (_Float16*)(ws + off); off = align256(off + (size_t)DMc * NQ * 2);
    _Float16* qbuf = (_Float16*)(ws + off); off = align256(off + (size_t)M * NQ  * 2);
    _Float16* kbuf = (_Float16*)(ws + off); off = align256(off + (size_t)M * NKV * 2);
    _Float16* vbuf = (_Float16*)(ws + off); off = align256(off + (size_t)M * NKV * 2);
    _Float16* vtb  = (_Float16*)(ws + off); off = align256(off + (size_t)M * NKV * 2);
    _Float16* obuf = xh;   // overlay

    auto blk = [](int n) { return (n + 255) / 256; };
    const dim3 tblk(32, 8);

    // 1) f16 conversion of x; LDS-tiled transposed f16 weights
    k_f32_to_f16<<<blk(M * DMc), 256, 0, stream>>>(x, xh, M * DMc);
    k_transpose_f16<<<dim3(NQ  / 32, DMc / 32), tblk, 0, stream>>>(wq, wqT, DMc, NQ);
    k_transpose_f16<<<dim3(NKV / 32, DMc / 32), tblk, 0, stream>>>(wk, wkT, DMc, NKV);
    k_transpose_f16<<<dim3(NKV / 32, DMc / 32), tblk, 0, stream>>>(wv, wvT, DMc, NKV);
    k_transpose_f16<<<dim3(DMc / 32, NQ  / 32), tblk, 0, stream>>>(wo, woT, NQ, DMc);

    // 2) QKV projections (WMMA)
    k_gemm_wmma<_Float16><<<dim3(NQ / 64,  M / 128), 128, 0, stream>>>(xh, wqT, qbuf, M, NQ,  DMc);
    k_gemm_wmma<_Float16><<<dim3(NKV / 64, M / 128), 128, 0, stream>>>(xh, wkT, kbuf, M, NKV, DMc);
    k_gemm_wmma<_Float16><<<dim3(NKV / 64, M / 128), 128, 0, stream>>>(xh, wvT, vbuf, M, NKV, DMc);

    // 3) RoPE on q and k; LDS-tiled transpose of v
    k_rope<<<blk(Bc * Sc * Hc  * 32), 256, 0, stream>>>(qbuf, cosb, sinb, Hc,  Bc * Sc * Hc  * 32);
    k_rope<<<blk(Bc * Sc * KHc * 32), 256, 0, stream>>>(kbuf, cosb, sinb, KHc, Bc * Sc * KHc * 32);
    k_vtrans<<<dim3(HDc / 32, Sc / 32, Bc * KHc), tblk, 0, stream>>>(vbuf, vtb);

    // 4) causal flash attention (WMMA)
    k_flash_attn<<<dim3(Sc / 64, Hc, Bc), 128, 0, stream>>>(qbuf, kbuf, vtb, obuf);

    // 5) output projection (WMMA, f32 out)
    k_gemm_wmma<float><<<dim3(DMc / 64, M / 128), 128, 0, stream>>>(obuf, woT, out, M, DMc, DMc);
}
